// SurfaceAbstraction_67997922230560
// MI455X (gfx1250) — matrix operand order repaired and checked by reading, not compiled, not run on hardware
//
#include <hip/hip_runtime.h>
#include <hip/hip_bf16.h>

// ---------------------------------------------------------------------------
// SurfaceAbstraction for MI455X (gfx1250, wave32, WMMA)
//
// 4-pass fused pipeline, one wave per point:
//   pass0: gather -> GEMM0 (bf16 WMMA) -> accumulate BN0 stats
//   pass1: gather -> GEMM0 -> BN0+ReLU -> GEMM1 -> BN1 stats
//   pass2: gather -> GEMM0..1 -> BN -> GEMM2 -> BN2 stats
//   pass3: gather -> GEMM0..2 -> BN2+ReLU -> max over 32 samples -> out
// Stats live in d_ws (512 floats). Activations never touch HBM.
// Cross-lane reductions use fire-and-forget LDS atomics (ds_add_f32 /
// ds_max_i32, no return -> no dscnt stalls). Gather rows are prefetched
// (global_prefetch) at kernel entry so the random L2 accesses overlap with
// weight staging.
// ---------------------------------------------------------------------------

#define N_PTS  40000
#define NS     32
#define CF     32
#define INCH   38           // 3 + 3 + CF
#define KPAD   64           // K padded for WMMA (38 -> 64)
#define C0     64
#define C1     64
#define C2     128
#define BN_EPS 1e-5f
#define WAVES  4            // waves (points) per block
#define TPB    (WAVES * 32)
#define XSTRIDE 72          // bf16 elems per sample row (64 + pad, 144B = 16B-mult)

typedef __bf16 bf16_t;
typedef __attribute__((ext_vector_type(16))) __bf16 v16bf;
typedef __attribute__((ext_vector_type(8)))  __bf16 v8bf;
typedef __attribute__((ext_vector_type(8)))  float  v8f;
typedef __attribute__((ext_vector_type(4)))  float  v4f;

union BFrag { v16bf v; v8bf h[2]; };

struct Params {
    const float* center;
    const float* normal;
    const float* feature;
    const int*   gidx;
    const float *W0, *b0, *g0, *be0;
    const float *W1, *b1, *g1, *be1;
    const float *W2, *b2, *g2, *be2;
    float* stats;   // ws: sum0[64] sq0[64] sum1[64] sq1[64] sum2[128] sq2[128]
    float* outF;    // d_out + 240000 : [N_PTS, C2]
};

__device__ __forceinline__ v8f wmma_bf16(v16bf a, v16bf b, v8f c) {
    return __builtin_amdgcn_wmma_f32_16x16x32_bf16(
        /*neg_a=*/false, a, /*neg_b=*/false, b,
        /*c_mod=*/(short)0, c, /*reuse_a=*/false, /*reuse_b=*/false);
}

// A-matrix fragment (16x32 bf16), ISA layout:
// lane<16: elems 0..7 = K k0+0..7,  elems 8..15 = K k0+16..23
// lane>=16: elems 0..7 = K k0+8..15, elems 8..15 = K k0+24..31
__device__ __forceinline__ v16bf load_a(const bf16_t* sw, int row, int half, int k0) {
    BFrag f;
    const bf16_t* p = sw + row * KPAD + k0 + half * 8;
    f.h[0] = *(const v8bf*)(p);
    f.h[1] = *(const v8bf*)(p + 16);
    return f.v;
}

// B-matrix fragment (32x16 bf16): lane holds column col = lane&15,
// same K layout as A: elems 0..7 = K k0+half*8+0..7, elems 8..15 = K k0+16+half*8+0..7.
__device__ __forceinline__ v16bf load_b(const bf16_t* sx, int col, int half, int k0) {
    BFrag f;
    const bf16_t* p = sx + col * XSTRIDE + k0 + half * 8;
    f.h[0] = *(const v8bf*)(p);
    f.h[1] = *(const v8bf*)(p + 16);
    return f.v;
}

// Intermediate layer: Y = W*X, then y = scale*y + shift, ReLU, write bf16 back
// into the same LDS slab (all reads complete before writeback; DS in-order).
template<int COUT>
__device__ __forceinline__ void layer_mid(const bf16_t* __restrict__ sw,
                                          bf16_t* __restrict__ sx,
                                          const float* __restrict__ sscale,
                                          const float* __restrict__ sshift,
                                          int lane) {
    constexpr int MT = COUT / 16;
    const int half = lane >> 4, l15 = lane & 15;
    v8f acc[MT][2];
#pragma unroll
    for (int mt = 0; mt < MT; ++mt) {
        const int row = mt * 16 + l15;
        v16bf a0 = load_a(sw, row, half, 0);
        v16bf a1 = load_a(sw, row, half, 32);
#pragma unroll
        for (int nt = 0; nt < 2; ++nt) {
            const int col = nt * 16 + l15;
            v16bf b0 = load_b(sx, col, half, 0);
            v16bf b1 = load_b(sx, col, half, 32);
            v8f c = {};
            c = wmma_bf16(a0, b0, c);
            c = wmma_bf16(a1, b1, c);
            acc[mt][nt] = c;
        }
    }
#pragma unroll
    for (int mt = 0; mt < MT; ++mt)
#pragma unroll
        for (int nt = 0; nt < 2; ++nt) {
            const int col = nt * 16 + l15;
            const int chb = mt * 16 + half * 8;
            v8bf o;
#pragma unroll
            for (int r = 0; r < 8; ++r) {
                const int ch = chb + r;
                float v = fmaxf(sscale[ch] * acc[mt][nt][r] + sshift[ch], 0.0f);
                o[r] = (bf16_t)v;
            }
            *(v8bf*)(sx + col * XSTRIDE + chb) = o;
        }
}

// Stats layer: y_pre = W*X + b ; accumulate per-channel sum / sum-of-squares.
// Per-lane register accumulation across the two N-tiles, then non-returning
// LDS float atomics (ds_add_f32) -- no dependent waits in the loop.
template<int COUT>
__device__ __forceinline__ void layer_stats(const bf16_t* __restrict__ sw,
                                            const bf16_t* __restrict__ sx,
                                            const float* __restrict__ sbias,
                                            float* __restrict__ ssum,
                                            float* __restrict__ ssq,
                                            int lane) {
    constexpr int MT = COUT / 16;
    const int half = lane >> 4, l15 = lane & 15;
#pragma unroll
    for (int mt = 0; mt < MT; ++mt) {
        const int row = mt * 16 + l15;
        const int chb = mt * 16 + half * 8;
        v16bf a0 = load_a(sw, row, half, 0);
        v16bf a1 = load_a(sw, row, half, 32);
        float s_part[8], q_part[8];
#pragma unroll
        for (int r = 0; r < 8; ++r) { s_part[r] = 0.0f; q_part[r] = 0.0f; }
#pragma unroll
        for (int nt = 0; nt < 2; ++nt) {
            const int col = nt * 16 + l15;
            v16bf b0 = load_b(sx, col, half, 0);
            v16bf b1 = load_b(sx, col, half, 32);
            v8f c = {};
            c = wmma_bf16(a0, b0, c);
            c = wmma_bf16(a1, b1, c);
#pragma unroll
            for (int r = 0; r < 8; ++r) {
                float v = c[r] + sbias[chb + r];
                s_part[r] += v;
                q_part[r] += v * v;
            }
        }
#pragma unroll
        for (int r = 0; r < 8; ++r) {
            atomicAdd(&ssum[chb + r], s_part[r]);
            atomicAdd(&ssq[chb + r], q_part[r]);
        }
    }
}

// Final layer: GEMM2 -> BN2+ReLU -> max over 32 samples -> global store.
// ReLU output is >= 0, so float max == integer max on the bit patterns:
// use non-returning ds_max_i32 into a wave-private LDS buffer (init 0),
// then 4 contiguous channels per lane -> one b128 LDS read + 16B global store.
__device__ __forceinline__ void layer_final(const bf16_t* __restrict__ sw,
                                            const bf16_t* __restrict__ sx,
                                            const float* __restrict__ sscale,
                                            const float* __restrict__ sshift,
                                            int* __restrict__ maxbuf,
                                            float* __restrict__ outF,
                                            int n, int lane) {
    const int half = lane >> 4, l15 = lane & 15;
#pragma unroll
    for (int i = 0; i < C2 / 32; ++i) maxbuf[lane + i * 32] = 0;  // wave-private
#pragma unroll
    for (int mt = 0; mt < C2 / 16; ++mt) {
        const int row = mt * 16 + l15;
        const int chb = mt * 16 + half * 8;
        v16bf a0 = load_a(sw, row, half, 0);
        v16bf a1 = load_a(sw, row, half, 32);
        float m[8];
#pragma unroll
        for (int nt = 0; nt < 2; ++nt) {
            v16bf b0 = load_b(sx, nt * 16 + l15, half, 0);
            v16bf b1 = load_b(sx, nt * 16 + l15, half, 32);
            v8f c = {};
            c = wmma_bf16(a0, b0, c);
            c = wmma_bf16(a1, b1, c);
#pragma unroll
            for (int r = 0; r < 8; ++r) {
                float v = fmaxf(sscale[chb + r] * c[r] + sshift[chb + r], 0.0f);
                m[r] = (nt == 0) ? v : fmaxf(m[r], v);
            }
        }
#pragma unroll
        for (int r = 0; r < 8; ++r)
            atomicMax(&maxbuf[chb + r], __float_as_int(m[r]));
    }
    // DS ops from the same wave are in-order: reads below see the atomics.
    const int ch0 = lane * 4;
    v4f o;
#pragma unroll
    for (int i = 0; i < 4; ++i) o[i] = __int_as_float(maxbuf[ch0 + i]);
    *(v4f*)(outF + (size_t)n * C2 + ch0) = o;
}

__device__ __forceinline__ void stage_weights(const float* __restrict__ W,
                                              bf16_t* __restrict__ sw,
                                              int cout, int cin) {
    for (int i = threadIdx.x; i < cout * KPAD; i += TPB) {
        const int co = i >> 6, k = i & 63;
        const float w = (k < cin) ? W[co * cin + k] : 0.0f;
        sw[i] = (bf16_t)w;
    }
}

__device__ __forceinline__ void stage_bn(const float* gsum, const float* gsq,
                                         const float* g, const float* be,
                                         const float* b, float* sc, float* sh,
                                         int C) {
    const float invCnt = 1.0f / (float)(N_PTS * NS);
    for (int ch = threadIdx.x; ch < C; ch += TPB) {
        const float mean  = gsum[ch] * invCnt;
        const float var   = gsq[ch] * invCnt - mean * mean;
        const float inv   = rsqrtf(var + BN_EPS);
        const float scale = g[ch] * inv;
        sc[ch] = scale;
        sh[ch] = be[ch] - scale * mean + scale * b[ch];  // bias folded in
    }
}

template<int PASS>
__global__ void __launch_bounds__(TPB)
surface_abstraction_pass(Params p) {
    __shared__ alignas(16) bf16_t sW0[C0 * KPAD];
    __shared__ alignas(16) bf16_t sW1[C1 * KPAD];
    __shared__ alignas(16) bf16_t sW2[C2 * KPAD];
    __shared__ alignas(16) bf16_t sX[WAVES][NS * XSTRIDE];
    __shared__ alignas(16) int    sMax[WAVES][C2];
    __shared__ float sc0[C0], sh0[C0], sc1[C1], sh1[C1], sc2[C2], sh2[C2];
    __shared__ float sbias[C2];
    __shared__ float ssum[C2], ssq[C2];

    // ---- start the random-gather dependency chain first --------------------
    // group_idx load + prefetch of the gathered rows overlaps with the weight
    // staging below (global_prefetch_b8; speculative, no counter, no VGPR).
    const int wave = threadIdx.x >> 5;
    const int lane = threadIdx.x & 31;
    const int n    = blockIdx.x * WAVES + wave;
    const int gi   = p.gidx[(size_t)n * NS + lane];
    __builtin_prefetch(p.feature + (size_t)gi * CF, 0, 3);
    __builtin_prefetch(p.center + gi * 3, 0, 3);
    __builtin_prefetch(p.normal + gi * 3, 0, 3);

    // ---- block-wide staging -------------------------------------------------
    stage_weights(p.W0, sW0, C0, INCH);
    if constexpr (PASS >= 1) stage_weights(p.W1, sW1, C1, C0);
    if constexpr (PASS >= 2) stage_weights(p.W2, sW2, C2, C1);

    if constexpr (PASS >= 1) stage_bn(p.stats + 0,   p.stats + 64,  p.g0, p.be0, p.b0, sc0, sh0, C0);
    if constexpr (PASS >= 2) stage_bn(p.stats + 128, p.stats + 192, p.g1, p.be1, p.b1, sc1, sh1, C1);
    if constexpr (PASS >= 3) stage_bn(p.stats + 256, p.stats + 384, p.g2, p.be2, p.b2, sc2, sh2, C2);

    if constexpr (PASS < 3) {
        constexpr int CT = (PASS == 2) ? C2 : C0;
        const float* bt = (PASS == 0) ? p.b0 : (PASS == 1) ? p.b1 : p.b2;
        for (int ch = threadIdx.x; ch < CT; ch += TPB) {
            sbias[ch] = bt[ch];
            ssum[ch] = 0.0f;
            ssq[ch]  = 0.0f;
        }
    }
    __syncthreads();

    // ---- gather: build X0 [KPAD x NS] bf16 (sample-major), one wave / point -
    bf16_t* sx = sX[wave];
    {
        const int s  = lane;
        const float cx = p.center[n * 3 + 0];
        const float cy = p.center[n * 3 + 1];
        const float cz = p.center[n * 3 + 2];
        bf16_t* xr = sx + s * XSTRIDE;
        xr[0] = (bf16_t)(p.center[gi * 3 + 0] - cx);
        xr[1] = (bf16_t)(p.center[gi * 3 + 1] - cy);
        xr[2] = (bf16_t)(p.center[gi * 3 + 2] - cz);
        xr[3] = (bf16_t)p.normal[gi * 3 + 0];
        xr[4] = (bf16_t)p.normal[gi * 3 + 1];
        xr[5] = (bf16_t)p.normal[gi * 3 + 2];
        const v4f* fp = (const v4f*)(p.feature + (size_t)gi * CF);
#pragma unroll
        for (int c4 = 0; c4 < CF / 4; ++c4) {
            v4f f = fp[c4];
#pragma unroll
            for (int j = 0; j < 4; ++j) xr[6 + c4 * 4 + j] = (bf16_t)f[j];
        }
#pragma unroll
        for (int k = INCH; k < KPAD; ++k) xr[k] = (bf16_t)0.0f;
    }
    // Wave-private slab: DS ops from the same wave are in-order; no barrier.

    // ---- fused layer pipeline ----------------------------------------------
    if constexpr (PASS == 0) {
        layer_stats<C0>(sW0, sx, sbias, ssum, ssq, lane);
    } else {
        layer_mid<C0>(sW0, sx, sc0, sh0, lane);
        if constexpr (PASS == 1) {
            layer_stats<C1>(sW1, sx, sbias, ssum, ssq, lane);
        } else {
            layer_mid<C1>(sW1, sx, sc1, sh1, lane);
            if constexpr (PASS == 2) {
                layer_stats<C2>(sW2, sx, sbias, ssum, ssq, lane);
            } else {
                layer_final(sW2, sx, sc2, sh2, sMax[wave], p.outF, n, lane);
            }
        }
    }

    // ---- flush block-local stats to global accumulators ---------------------
    if constexpr (PASS < 3) {
        __syncthreads();
        constexpr int CT = (PASS == 2) ? C2 : C0;
        float* gdst = p.stats + (PASS == 0 ? 0 : PASS == 1 ? 128 : 256);
        for (int ch = threadIdx.x; ch < CT; ch += TPB) {
            atomicAdd(&gdst[ch],      ssum[ch]);
            atomicAdd(&gdst[ch + CT], ssq[ch]);
        }
    }
}

__global__ void init_ws_kernel(float* ws) {
    const int i = threadIdx.x;
    if (i < 512) ws[i] = 0.0f;
}

// center -> out[0:120000], normal -> out[120000:240000], offset bits at end.
__global__ void copy_misc_kernel(const float* __restrict__ c,
                                 const float* __restrict__ nrm,
                                 const int* __restrict__ off,
                                 float* __restrict__ out) {
    const int i = blockIdx.x * blockDim.x + threadIdx.x;
    if (i < N_PTS * 3) {
        out[i] = c[i];
        out[N_PTS * 3 + i] = nrm[i];
    }
    if (i == 0) {
        ((int*)out)[2 * N_PTS * 3 + (size_t)N_PTS * C2] = off[0];
    }
}

extern "C" void kernel_launch(void* const* d_in, const int* in_sizes, int n_in,
                              void* d_out, int out_size, void* d_ws, size_t ws_size,
                              hipStream_t stream) {
    Params p;
    p.center  = (const float*)d_in[0];
    p.normal  = (const float*)d_in[1];
    p.feature = (const float*)d_in[2];
    const int* offset = (const int*)d_in[3];
    p.gidx    = (const int*)d_in[4];
    p.W0 = (const float*)d_in[5];  p.b0 = (const float*)d_in[6];
    p.g0 = (const float*)d_in[7];  p.be0 = (const float*)d_in[8];
    p.W1 = (const float*)d_in[9];  p.b1 = (const float*)d_in[10];
    p.g1 = (const float*)d_in[11]; p.be1 = (const float*)d_in[12];
    p.W2 = (const float*)d_in[13]; p.b2 = (const float*)d_in[14];
    p.g2 = (const float*)d_in[15]; p.be2 = (const float*)d_in[16];
    p.stats = (float*)d_ws;
    p.outF  = (float*)d_out + 2 * N_PTS * 3;

    const int nblk = N_PTS / WAVES;

    init_ws_kernel<<<1, 512, 0, stream>>>((float*)d_ws);
    copy_misc_kernel<<<(N_PTS * 3 + 255) / 256, 256, 0, stream>>>(
        p.center, p.normal, offset, (float*)d_out);

    surface_abstraction_pass<0><<<nblk, TPB, 0, stream>>>(p);
    surface_abstraction_pass<1><<<nblk, TPB, 0, stream>>>(p);
    surface_abstraction_pass<2><<<nblk, TPB, 0, stream>>>(p);
    surface_abstraction_pass<3><<<nblk, TPB, 0, stream>>>(p);
}
